// PalmSelfAttention_31971736551997
// MI455X (gfx1250) — matrix operand order repaired and checked by reading, not compiled, not run on hardware
//
#include <hip/hip_runtime.h>
#include <hip/hip_bf16.h>
#include <cstdint>

#define BB  2
#define SS  2048
#define HH  1024
#define NHH 16
#define HD  64

#if defined(__has_builtin)
#if __has_builtin(__builtin_amdgcn_sched_barrier)
#define SCHED_BARRIER() __builtin_amdgcn_sched_barrier(0)
#endif
#endif
#ifndef SCHED_BARRIER
#define SCHED_BARRIER()
#endif

typedef __attribute__((ext_vector_type(16))) __bf16 v16bf;
typedef __attribute__((ext_vector_type(8)))  float  v8f;

union Frag { v16bf v; uint4 q[2]; };

__device__ __forceinline__ unsigned short f2bf(float f) {
    unsigned u = __float_as_uint(f);
    u += 0x7FFFu + ((u >> 16) & 1u);          // round-to-nearest-even
    return (unsigned short)(u >> 16);
}

__device__ __forceinline__ v8f v8f_zero() {
    v8f z;
    #pragma unroll
    for (int i = 0; i < 8; ++i) z[i] = 0.0f;
    return z;
}

// ---------------------------------------------------------------- fp32 -> bf16 (8 elems/thread)
__global__ void cvt_bf16(const float* __restrict__ src,
                         unsigned short* __restrict__ dst, int n) {
    int i = (blockIdx.x * blockDim.x + threadIdx.x) * 8;
    if (i >= n) return;
    float4 f0 = *reinterpret_cast<const float4*>(src + i);
    float4 f1 = *reinterpret_cast<const float4*>(src + i + 4);
    union { unsigned short s[8]; uint4 u; } o;
    o.s[0] = f2bf(f0.x); o.s[1] = f2bf(f0.y); o.s[2] = f2bf(f0.z); o.s[3] = f2bf(f0.w);
    o.s[4] = f2bf(f1.x); o.s[5] = f2bf(f1.y); o.s[6] = f2bf(f1.z); o.s[7] = f2bf(f1.w);
    *reinterpret_cast<uint4*>(dst + i) = o.u;
}

// ---------------------------------------------------------------- QKV GEMM (double buffered)
// mixed[m][n] = sum_k hidden[m][k] * qkv_w[n][k] + bias[n]
// M = B*S = 4096, N = 3H = 3072, K = H = 1024
// grid (M/16, N/512), block 256 (8 waves); wave owns a 16x64 tile.
__global__ __launch_bounds__(256)
void qkv_gemm(const unsigned short* __restrict__ hid,  // bf16 [4096,1024]
              const unsigned short* __restrict__ w,    // bf16 [3072,1024]
              const float* __restrict__ bias,          // f32  [3072]
              unsigned short* __restrict__ qws,        // bf16 [2,16,2048,64] (pre-scaled by 1/8)
              unsigned short* __restrict__ kws,        // bf16 [2,16,2048,64]
              unsigned short* __restrict__ vws)        // bf16 [2,16,64,2048] (transposed)
{
    const int lane = threadIdx.x & 31;
    const int wv   = threadIdx.x >> 5;
    const int m0   = blockIdx.x * 16;
    const int n0   = blockIdx.y * 512 + wv * 64;
    const int mrow = lane & 15;
    const int half = lane >> 4;

    v8f acc[4];
    #pragma unroll
    for (int t = 0; t < 4; ++t) acc[t] = v8f_zero();

    const unsigned short* arow = hid + (size_t)(m0 + mrow) * HH;

    auto loadAB = [&](Frag& a, Frag (&b)[4], int k0) {
        a.q[0] = *reinterpret_cast<const uint4*>(arow + k0 + 16 * half);
        a.q[1] = *reinterpret_cast<const uint4*>(arow + k0 + 16 * half + 8);
        #pragma unroll
        for (int t = 0; t < 4; ++t) {
            const unsigned short* bp =
                w + (size_t)(n0 + t * 16 + mrow) * HH + k0 + 16 * half;
            b[t].q[0] = *reinterpret_cast<const uint4*>(bp);
            b[t].q[1] = *reinterpret_cast<const uint4*>(bp + 8);
        }
    };
    auto mac = [&](Frag& a, Frag (&b)[4]) {
        #pragma unroll
        for (int t = 0; t < 4; ++t)
            acc[t] = __builtin_amdgcn_wmma_f32_16x16x32_bf16(
                         false, a.v, false, b[t].v, (short)0, acc[t], false, false);
    };

    Frag a0, a1, b0[4], b1[4];
    loadAB(a0, b0, 0);
    for (int k0 = 0; k0 < HH; k0 += 64) {
        loadAB(a1, b1, k0 + 32);        // next buffer in flight...
        SCHED_BARRIER();                // ...keep its loads above the WMMAs
        mac(a0, b0);
        SCHED_BARRIER();
        const int kn = (k0 + 64 < HH) ? k0 + 64 : 0;   // clamped (redundant last load)
        loadAB(a0, b0, kn);
        SCHED_BARRIER();
        mac(a1, b1);
        SCHED_BARRIER();
    }

    // Epilogue: C layout lane = (n = lane%16, vgpr r -> m = r + 8*half)
    #pragma unroll
    for (int t = 0; t < 4; ++t) {
        const int n = n0 + t * 16 + mrow;
        const int which = n >> 10;      // 0=Q 1=K 2=V (uniform per block)
        const int c = n & 1023;
        const int h = c >> 6;
        const int d = c & 63;
        const float bv = bias[n];
        #pragma unroll
        for (int r = 0; r < 8; ++r) {
            const int mg = m0 + r + 8 * half;
            const int b  = mg >> 11;
            const int s  = mg & 2047;
            const size_t bh = (size_t)(b * NHH + h);
            const float val = acc[t][r] + bv;
            if (which == 0)      qws[(bh * SS + s) * HD + d] = f2bf(val * 0.125f);
            else if (which == 1) kws[(bh * SS + s) * HD + d] = f2bf(val);
            else                 vws[(bh * HD + d) * SS + s] = f2bf(val);
        }
    }
}

// ---------------------------------------------------------------- flash attention (double buffered)
// grid (S/128, B*NH), block 256 (8 waves); wave owns one 16-row Q tile.
__global__ __launch_bounds__(256)
void flash_attn(const unsigned short* __restrict__ qws,
                const unsigned short* __restrict__ kws,
                const unsigned short* __restrict__ vws,
                const float* __restrict__ mask,
                float* __restrict__ out)
{
    __shared__ unsigned short lds[8][16 * 32];   // per-wave 16x32 bf16 P tile

    const int lane = threadIdx.x & 31;
    const int wv   = threadIdx.x >> 5;
    const int mrow = lane & 15;
    const int half = lane >> 4;
    const int bh   = blockIdx.y;
    const int b    = bh >> 4;
    const int h    = bh & 15;
    const int q0   = (blockIdx.x * 8 + wv) * 16;

    const unsigned short* qbase = qws + (size_t)bh * SS * HD;
    const unsigned short* kbase = kws + (size_t)bh * SS * HD;
    const unsigned short* vbase = vws + (size_t)bh * HD * SS;
    const float*          mbase = mask + ((size_t)b * SS + q0) * SS;

    // Q A-frags for both 32-wide d-tiles, resident for the whole loop.
    Frag qa[2];
    #pragma unroll
    for (int kt = 0; kt < 2; ++kt) {
        const unsigned short* qp = qbase + (size_t)(q0 + mrow) * HD + kt * 32;
        qa[kt].q[0] = *reinterpret_cast<const uint4*>(qp + 16 * half);
        qa[kt].q[1] = *reinterpret_cast<const uint4*>(qp + 16 * half + 8);
    }

    float m_i[8], l_i[8];
    #pragma unroll
    for (int r = 0; r < 8; ++r) { m_i[r] = -3.0e38f; l_i[r] = 0.0f; }
    v8f acc[4];
    #pragma unroll
    for (int t = 0; t < 4; ++t) acc[t] = v8f_zero();

    unsigned short* myld = lds[wv];

    // B-frags for a 32-key chunk: [key-group f][d-tile]
    auto loadK = [&](Frag (&kb)[2][2], int kc) {
        #pragma unroll
        for (int f = 0; f < 2; ++f) {
            const unsigned short* kp =
                kbase + (size_t)(kc + f * 16 + mrow) * HD + 16 * half;
            kb[f][0].q[0] = *reinterpret_cast<const uint4*>(kp);
            kb[f][0].q[1] = *reinterpret_cast<const uint4*>(kp + 8);
            kb[f][1].q[0] = *reinterpret_cast<const uint4*>(kp + 32);
            kb[f][1].q[1] = *reinterpret_cast<const uint4*>(kp + 40);
        }
    };

    auto attn_step = [&](Frag (&kb)[2][2], int kc) {
        // V frags + mask rows issued up-front: they overlap score WMMAs + softmax.
        Frag vb[4];
        #pragma unroll
        for (int t = 0; t < 4; ++t) {
            const unsigned short* vp =
                vbase + (size_t)(t * 16 + mrow) * SS + kc + 16 * half;
            vb[t].q[0] = *reinterpret_cast<const uint4*>(vp);
            vb[t].q[1] = *reinterpret_cast<const uint4*>(vp + 8);
        }
        float mval[2][8];
        #pragma unroll
        for (int f = 0; f < 2; ++f)
            #pragma unroll
            for (int r = 0; r < 8; ++r)
                mval[f][r] = mbase[(size_t)(r + 8 * half) * SS + kc + f * 16 + mrow];
        // prefetch mask rows a few chunks ahead (streamed 33.5MB array)
        __builtin_prefetch(mbase + (size_t)(8 * half) * SS + kc + 256, 0, 1);
        SCHED_BARRIER();   // keep V/mask loads above the score WMMAs

        // ---- scores: two 16-key groups, each = 2 chained WMMAs over d
        v8f sc[2];
        #pragma unroll
        for (int f = 0; f < 2; ++f) {
            v8f z = v8f_zero();
            z = __builtin_amdgcn_wmma_f32_16x16x32_bf16(
                    false, qa[0].v, false, kb[f][0].v, (short)0, z, false, false);
            z = __builtin_amdgcn_wmma_f32_16x16x32_bf16(
                    false, qa[1].v, false, kb[f][1].v, (short)0, z, false, false);
            sc[f] = z;
            #pragma unroll
            for (int r = 0; r < 8; ++r) sc[f][r] += mval[f][r];
        }

        // ---- online softmax (row reductions over the 16-lane N group)
        #pragma unroll
        for (int r = 0; r < 8; ++r) {
            float v = fmaxf(sc[0][r], sc[1][r]);
            #pragma unroll
            for (int x = 1; x < 16; x <<= 1) v = fmaxf(v, __shfl_xor(v, x, 32));
            const float nm = fmaxf(m_i[r], v);
            const float sscale = __expf(m_i[r] - nm);
            m_i[r] = nm;
            const float p0 = __expf(sc[0][r] - nm);
            const float p1 = __expf(sc[1][r] - nm);
            sc[0][r] = p0; sc[1][r] = p1;
            float s2 = p0 + p1;
            #pragma unroll
            for (int x = 1; x < 16; x <<= 1) s2 += __shfl_xor(s2, x, 32);
            l_i[r] = l_i[r] * sscale + s2;
            #pragma unroll
            for (int t = 0; t < 4; ++t) acc[t][r] *= sscale;
        }

        // ---- P: C-layout -> LDS -> A-layout
        #pragma unroll
        for (int f = 0; f < 2; ++f)
            #pragma unroll
            for (int r = 0; r < 8; ++r)
                myld[(r + 8 * half) * 32 + f * 16 + mrow] = f2bf(sc[f][r]);
        asm volatile("s_wait_dscnt 0x0" ::: "memory");
        Frag pa;
        pa.q[0] = *reinterpret_cast<const uint4*>(myld + mrow * 32 + 16 * half);
        pa.q[1] = *reinterpret_cast<const uint4*>(myld + mrow * 32 + 16 * half + 8);
        asm volatile("s_wait_dscnt 0x0" ::: "memory");

        // ---- ctx += P @ V
        #pragma unroll
        for (int t = 0; t < 4; ++t)
            acc[t] = __builtin_amdgcn_wmma_f32_16x16x32_bf16(
                         false, pa.v, false, vb[t].v, (short)0, acc[t], false, false);
    };

    // Two-phase software pipeline over 32-key chunks.
    Frag kbA[2][2], kbB[2][2];
    loadK(kbA, 0);
    for (int kc = 0; kc < SS; kc += 64) {
        loadK(kbB, kc + 32);
        SCHED_BARRIER();            // next K chunk stays above current step
        attn_step(kbA, kc);
        const int kn = (kc + 64 < SS) ? kc + 64 : 0;   // clamped (redundant last load)
        loadK(kbA, kn);
        SCHED_BARRIER();
        attn_step(kbB, kc + 32);
    }

    // ---- normalize + store (coalesced f32)
    #pragma unroll
    for (int r = 0; r < 8; ++r) {
        const float inv = 1.0f / l_i[r];
        const int s = q0 + r + 8 * half;
        float* orow = out + ((size_t)b * SS + s) * HH + h * HD;
        #pragma unroll
        for (int t = 0; t < 4; ++t)
            orow[t * 16 + mrow] = acc[t][r] * inv;
    }
}

// ----------------------------------------------------------------
extern "C" void kernel_launch(void* const* d_in, const int* in_sizes, int n_in,
                              void* d_out, int out_size, void* d_ws, size_t ws_size,
                              hipStream_t stream) {
    (void)in_sizes; (void)n_in; (void)out_size; (void)ws_size;
    const float* hidden = (const float*)d_in[0];
    const float* mask   = (const float*)d_in[1];
    const float* qkv_w  = (const float*)d_in[2];
    const float* qkv_b  = (const float*)d_in[3];
    float* out = (float*)d_out;

    unsigned short* hid_bf = (unsigned short*)d_ws;                    // 4 Mi elems
    unsigned short* w_bf   = hid_bf + (size_t)BB * SS * HH;            // 3 Mi elems
    unsigned short* qws    = w_bf   + (size_t)3 * HH * HH;             // 4 Mi elems
    unsigned short* kws    = qws    + (size_t)BB * NHH * SS * HD;
    unsigned short* vws    = kws    + (size_t)BB * NHH * SS * HD;      // total ~40 MB

    const int nh = BB * SS * HH;
    const int nw = 3 * HH * HH;
    cvt_bf16<<<(nh / 8 + 255) / 256, 256, 0, stream>>>(hidden, hid_bf, nh);
    cvt_bf16<<<(nw / 8 + 255) / 256, 256, 0, stream>>>(qkv_w, w_bf, nw);

    qkv_gemm<<<dim3(BB * SS / 16, 3 * HH / 512), 256, 0, stream>>>(
        hid_bf, w_bf, qkv_b, qws, kws, vws);

    flash_attn<<<dim3(SS / 128, BB * NHH), 256, 0, stream>>>(
        qws, kws, vws, mask, out);
}
